// LatticeQuantizedParameter_37211596652925
// MI455X (gfx1250) — matrix operand order repaired and checked by reading, not compiled, not run on hardware
//
#include <hip/hip_runtime.h>

#define M_OUT   11008
#define N_IN    4096
#define GPR     (N_IN / 8)      // 512 index groups per weight row
#define TILE_B  128
#define TILE_M  128
#define KCHUNK  64              // staged K per barrier (2 WMMA k-steps)
#define LDSTR   72              // bf16 elems per LDS row (144B = 36 banks -> conflict free)

#define SCALE_F      0.9f
#define INV_RESID_F  (1.0f / 2.04f)

typedef __attribute__((ext_vector_type(16))) __bf16 v16bf;
typedef __attribute__((ext_vector_type(8)))  __bf16 v8bf;
typedef __attribute__((ext_vector_type(8)))  float  v8f;
typedef __attribute__((ext_vector_type(4)))  float  v4f;

__global__ __launch_bounds__(256)
void lattice_rvq_gemm_wmma(const float* __restrict__ x,
                           const int*   __restrict__ idxs0,
                           const int*   __restrict__ idxs1,
                           const float* __restrict__ cb,
                           float*       __restrict__ out)
{
    __shared__ __bf16 x_lds[2][TILE_B * LDSTR];   // 2 x 18432 B
    __shared__ __bf16 w_lds[2][TILE_M * LDSTR];   // 2 x 18432 B
    __shared__ float  cb_lds[256 * 8];            //     8192 B

    const int tid    = threadIdx.x;
    const int lane   = tid & 31;
    const int wave   = tid >> 5;       // 0..7
    const int wave_t = wave & 1;       // token half   (64 tokens each)
    const int wave_r = wave >> 1;      // row quarter  (32 rows each)

    const int tokenBase = blockIdx.x * TILE_B;
    const int rowBase   = blockIdx.y * TILE_M;

    // ---- codebook -> LDS (256 x 8 f32), 8 floats per thread ----
    {
        const v4f* src = (const v4f*)cb;
        v4f*       dst = (v4f*)cb_lds;
        dst[tid * 2 + 0] = src[tid * 2 + 0];
        dst[tid * 2 + 1] = src[tid * 2 + 1];
    }
    __syncthreads();   // cb_lds visible before first decode

    // ---- accumulators: 4 token-tiles x 2 row-tiles of 16x16 f32 ----
    v8f c[4][2];
    #pragma unroll
    for (int i = 0; i < 4; ++i)
        #pragma unroll
        for (int j = 0; j < 2; ++j)
            #pragma unroll
            for (int e = 0; e < 8; ++e)
                c[i][j][e] = 0.0f;

    const int xtok  = tid >> 1;   // 0..127 : token row staged by this thread
    const int xhalf = tid & 1;    // which 32-float half of the 64-wide K chunk

    const int klo = (lane >> 4) * 8;    // A-fragment K offset per ISA layout
    const int kb  = (lane >> 4) * 8;    // B-fragment K offset per ISA layout
    const int ln  = lane & 15;

    // ---------- staging helper (to buffer `buf`, K chunk at `k0`) ----------
    auto stage = [&](int buf, int k0) {
        // x tile: 128 tokens x 64 K, f32 -> bf16 (32 floats per thread)
        {
            const float* xg = x + (size_t)(tokenBase + xtok) * N_IN + k0 + xhalf * 32;
            __bf16* dst = &x_lds[buf][xtok * LDSTR + xhalf * 32];
            #pragma unroll
            for (int q = 0; q < 4; ++q) {
                const v4f f0 = ((const v4f*)xg)[2 * q + 0];
                const v4f f1 = ((const v4f*)xg)[2 * q + 1];
                v8bf bv;
                #pragma unroll
                for (int j = 0; j < 4; ++j) {
                    bv[j]     = (__bf16)f0[j];
                    bv[4 + j] = (__bf16)f1[j];
                }
                *(v8bf*)&dst[q * 8] = bv;
            }
        }
        // W tile: 128 rows x 64 K = 8 groups/row; 4 groups per thread
        #pragma unroll
        for (int s = 0; s < 4; ++s) {
            const int gidx = tid + s * 256;      // 0..1023
            const int rl   = gidx >> 3;          // 0..127 : local row
            const int gs   = gidx & 7;           // 0..7   : group within K chunk
            const int m    = rowBase + rl;
            const int g    = (k0 >> 3) + gs;
            const int i0   = idxs0[(size_t)m * GPR + g];
            const int i1   = idxs1[(size_t)m * GPR + g];
            const v4f* c0  = (const v4f*)&cb_lds[i0 * 8];
            const v4f* c1  = (const v4f*)&cb_lds[i1 * 8];
            const v4f wlo  = c0[0] + c1[0] * INV_RESID_F;
            const v4f whi  = c0[1] + c1[1] * INV_RESID_F;
            v8bf wb;
            #pragma unroll
            for (int j = 0; j < 4; ++j) {
                wb[j]     = (__bf16)wlo[j];
                wb[4 + j] = (__bf16)whi[j];
            }
            *(v8bf*)&w_lds[buf][rl * LDSTR + gs * 8] = wb;
        }
    };

    // ---------- software pipeline: prologue ----------
    stage(0, 0);
    __syncthreads();

    for (int k0 = 0; k0 < N_IN; k0 += KCHUNK) {
        const int cur = (k0 >> 6) & 1;
        const int nxt = cur ^ 1;

        // issue next chunk's global loads + decode first (hidden under WMMAs)
        if (k0 + KCHUNK < N_IN)
            stage(nxt, k0 + KCHUNK);

        // ---- compute on current buffer: 2 k-halves x 8 WMMA ----
        #pragma unroll
        for (int kk = 0; kk < KCHUNK; kk += 32) {
            v16bf a[4], b[2];
            #pragma unroll
            for (int tt = 0; tt < 4; ++tt) {
                const int tok = wave_t * 64 + tt * 16 + ln;   // A: lane%16 = M (token)
                const v8bf lo = *(const v8bf*)&x_lds[cur][tok * LDSTR + kk + klo];
                const v8bf hi = *(const v8bf*)&x_lds[cur][tok * LDSTR + kk + klo + 16];
                #pragma unroll
                for (int j = 0; j < 8; ++j) { a[tt][j] = lo[j]; a[tt][8 + j] = hi[j]; }
            }
            #pragma unroll
            for (int rt = 0; rt < 2; ++rt) {
                const int r   = wave_r * 32 + rt * 16 + ln;   // B: lane%16 = N (weight row)
                const v8bf lo = *(const v8bf*)&w_lds[cur][r * LDSTR + kk + kb];
                const v8bf hi = *(const v8bf*)&w_lds[cur][r * LDSTR + kk + kb + 16];
                #pragma unroll
                for (int j = 0; j < 8; ++j) { b[rt][j] = lo[j]; b[rt][8 + j] = hi[j]; }
            }
            #pragma unroll
            for (int tt = 0; tt < 4; ++tt)
                #pragma unroll
                for (int rt = 0; rt < 2; ++rt)
                    c[tt][rt] = __builtin_amdgcn_wmma_f32_16x16x32_bf16(
                        false, a[tt], false, b[rt], (short)0, c[tt][rt], false, false);
        }

        // publish next buffer / retire reads of current buffer
        __syncthreads();
    }

    // ---- store: D layout -> lane%16 = N (contiguous out features) ----
    #pragma unroll
    for (int tt = 0; tt < 4; ++tt) {
        #pragma unroll
        for (int rt = 0; rt < 2; ++rt) {
            const int token = tokenBase + wave_t * 64 + tt * 16 + (lane >> 4) * 8;
            const int col   = rowBase + wave_r * 32 + rt * 16 + ln;
            float* o = out + (size_t)token * M_OUT + col;
            #pragma unroll
            for (int r = 0; r < 8; ++r)
                o[(size_t)r * M_OUT] = c[tt][rt][r] * SCALE_F;
        }
    }
}

extern "C" void kernel_launch(void* const* d_in, const int* in_sizes, int n_in,
                              void* d_out, int out_size, void* d_ws, size_t ws_size,
                              hipStream_t stream) {
    (void)in_sizes; (void)n_in; (void)d_ws; (void)ws_size; (void)out_size;
    const float* x     = (const float*)d_in[0];
    const int*   idxs0 = (const int*)d_in[1];
    const int*   idxs1 = (const int*)d_in[2];
    const float* cb    = (const float*)d_in[3];
    float*       out   = (float*)d_out;

    dim3 grid(4096 / TILE_B, M_OUT / TILE_M);   // 32 x 86
    dim3 block(256);
    lattice_rvq_gemm_wmma<<<grid, block, 0, stream>>>(x, idxs0, idxs1, cb, out);
}
